// ProbabiltyFlow_8933531976198
// MI455X (gfx1250) — matrix operand order repaired and checked by reading, not compiled, not run on hardware
//
#include <hip/hip_runtime.h>

// ---------------------------------------------------------------------------
// ProbabilityFlow fused kernel for MI455X (gfx1250, wave32, WMMA)
//   dz_dt   = k*z + k*MLP([z, t])
//   dlogp   = -(2k + k * sum_i W3[i,:] @ u2_i)     (exact Jacobian trace)
// Layer-2 (512x512) as bf16 WMMA / f32 accumulate. 32 batch rows per WG,
// 64 N-columns per wave: each A fragment feeds 4 WMMAs, each W2 B fragment
// feeds 6 WMMAs -> ~256B LDS per WMMA (at the WGP LDS-bandwidth roofline).
// LDS layout interleaved [row][mat][col] so all 6 A fragments per k-step are
// reachable with 16-bit DS immediate offsets from one per-lane base.
// ---------------------------------------------------------------------------

typedef __attribute__((ext_vector_type(16))) __bf16 v16bf;
typedef __attribute__((ext_vector_type(8)))  float  v8f;

#define B_TOTAL 131072
#define INNER   512
#define KFLOW   1.0f
#define MROWS   32    // batch rows per workgroup (2 WMMA M-tiles)
#define ASTRIDE 520   // 512 + 8 halves padding per (row,mat) line
#define RSTRIDE (3 * ASTRIDE)   // halves per row (3 interleaved matrices)

__device__ __forceinline__ unsigned short f2bf(float x) {
    unsigned int u = __float_as_uint(x);
    unsigned int r = u + 0x7FFFu + ((u >> 16) & 1u);   // round-to-nearest-even
    return (unsigned short)(r >> 16);
}

// ---- prep: W2 f32 -> bf16 (L2-resident, reused by every workgroup) --------
__global__ void cvt_w2_bf16(const float* __restrict__ w2,
                            unsigned short* __restrict__ o) {
    int i = blockIdx.x * blockDim.x + threadIdx.x;   // grid sized exactly
    o[i] = f2bf(w2[i]);
}

// ---- main fused kernel ----------------------------------------------------
__global__ __launch_bounds__(256)
void pflow_kernel(const float* __restrict__ t_p,
                  const float* __restrict__ z,
                  const float* __restrict__ W1,
                  const float* __restrict__ b1,
                  const unsigned short* __restrict__ W2bf,
                  const float* __restrict__ b2,
                  const float* __restrict__ W3,
                  const float* __restrict__ b3,
                  float* __restrict__ out)
{
    __shared__ unsigned short As[MROWS * RSTRIDE];   // [row][h1|u1_0|u1_1]
    __shared__ float red[MROWS][4];                  // y0,y1,t0,t1 per row

    const int tid = threadIdx.x;
    const int r0  = blockIdx.x * MROWS;
    const float tval = t_p[0];

    if (tid < MROWS * 4) ((float*)red)[tid] = 0.0f;

    // -------- Phase 1: layer 1 + ReLU-gated tangent seeds (VALU) ----------
    {
        const int row = tid >> 3;          // 0..31 : batch row in tile
        const int sub = tid & 7;           // 8 threads cooperate per row
        const float z0 = z[(r0 + row) * 2 + 0];
        const float z1 = z[(r0 + row) * 2 + 1];
        unsigned short* a0 = As + row * RSTRIDE;     // h1
        unsigned short* a1 = a0 + ASTRIDE;           // u1_0
        unsigned short* a2 = a0 + 2 * ASTRIDE;       // u1_1
#pragma unroll 4
        for (int jj = 0; jj < 64; ++jj) {
            const int j = jj * 8 + sub;              // hidden unit
            const float4 w = reinterpret_cast<const float4*>(W1)[j];
            float pre = fmaf(w.x, z0, fmaf(w.y, z1, fmaf(w.z + w.w, tval, b1[j])));
            const bool m = pre > 0.0f;
            a0[j] = f2bf(m ? pre : 0.0f);
            a1[j] = m ? f2bf(w.x) : (unsigned short)0;
            a2[j] = m ? f2bf(w.y) : (unsigned short)0;
        }
    }
    __syncthreads();

    // -------- Phase 2: layer 2 as bf16 WMMA, K = 512 ----------------------
    const int wave  = tid >> 5;            // 0..7 -> 64-column N slice
    const int lane  = tid & 31;
    const int nbase = wave * 64;
    const int mrow  = lane & 15;           // A row / B column within tile
    const int khalf = lane >> 4;           // K-half select per ISA layout

    union AV { v16bf v; uint4 q[2]; };

    v8f acc[3][2][4];                      // [mat][m-tile][n-tile]
#pragma unroll
    for (int a = 0; a < 3; ++a)
#pragma unroll
        for (int mt = 0; mt < 2; ++mt)
#pragma unroll
            for (int nt = 0; nt < 4; ++nt)
#pragma unroll
                for (int e = 0; e < 8; ++e) acc[a][mt][nt][e] = 0.f;

    // per-lane LDS base: all 6 fragments of a k-step are at
    //   abase + (mt*16)*RSTRIDE + mat*ASTRIDE + kb   (+16 for high K half)
    // max offset ~53KB -> fits the 16-bit DS immediate.
    const unsigned short* abase = As + mrow * RSTRIDE + khalf * 8;

#pragma unroll 2
    for (int kb = 0; kb < INNER; kb += 32) {
        AV af[3][2];                       // [mat][m-tile]
#pragma unroll
        for (int mt = 0; mt < 2; ++mt) {
#pragma unroll
            for (int a = 0; a < 3; ++a) {
                const unsigned short* p =
                    abase + mt * 16 * RSTRIDE + a * ASTRIDE + kb;
                af[a][mt].q[0] = *reinterpret_cast<const uint4*>(p);
                af[a][mt].q[1] = *reinterpret_cast<const uint4*>(p + 16);
            }
        }
#pragma unroll
        for (int nt = 0; nt < 4; ++nt) {
            const int ncol = nbase + nt * 16 + mrow;
            const unsigned short* bp = W2bf + ncol * INNER + kb + khalf * 8;
            AV bf;
            bf.q[0] = *reinterpret_cast<const uint4*>(bp);
            bf.q[1] = *reinterpret_cast<const uint4*>(bp + 16);
            // one B fragment feeds 6 WMMAs (3 matrices x 2 M-tiles)
#pragma unroll
            for (int mt = 0; mt < 2; ++mt) {
                acc[0][mt][nt] = __builtin_amdgcn_wmma_f32_16x16x32_bf16(
                    false, af[0][mt].v, false, bf.v, (short)0,
                    acc[0][mt][nt], false, false);
                acc[1][mt][nt] = __builtin_amdgcn_wmma_f32_16x16x32_bf16(
                    false, af[1][mt].v, false, bf.v, (short)0,
                    acc[1][mt][nt], false, false);
                acc[2][mt][nt] = __builtin_amdgcn_wmma_f32_16x16x32_bf16(
                    false, af[2][mt].v, false, bf.v, (short)0,
                    acc[2][mt][nt], false, false);
            }
        }
    }

    // -------- Epilogue: bias + ReLU mask + W3 projection ------------------
#pragma unroll
    for (int mt = 0; mt < 2; ++mt) {
        float py0[8], py1[8], pt0[8], pt1[8];
#pragma unroll
        for (int i = 0; i < 8; ++i) { py0[i] = py1[i] = pt0[i] = pt1[i] = 0.f; }

#pragma unroll
        for (int nt = 0; nt < 4; ++nt) {
            const int n = nbase + nt * 16 + mrow;
            const float b2n = b2[n];
            const float w30 = W3[n];            // W3[0, n]
            const float w31 = W3[INNER + n];    // W3[1, n]
#pragma unroll
            for (int i = 0; i < 8; ++i) {
                const float pre = acc[0][mt][nt][i] + b2n;
                const bool  m   = pre > 0.0f;
                const float h2  = m ? pre : 0.0f;
                const float q0  = m ? acc[1][mt][nt][i] : 0.0f;
                const float q1  = m ? acc[2][mt][nt][i] : 0.0f;
                py0[i] = fmaf(w30, h2, py0[i]);
                py1[i] = fmaf(w31, h2, py1[i]);
                pt0[i] = fmaf(w30, q0, pt0[i]);
                pt1[i] = fmaf(w31, q1, pt1[i]);
            }
        }

        // butterfly reduce over the 16 lanes sharing the same rows
#pragma unroll
        for (int off = 1; off <= 8; off <<= 1) {
#pragma unroll
            for (int i = 0; i < 8; ++i) {
                py0[i] += __shfl_xor(py0[i], off, 32);
                py1[i] += __shfl_xor(py1[i], off, 32);
                pt0[i] += __shfl_xor(pt0[i], off, 32);
                pt1[i] += __shfl_xor(pt1[i], off, 32);
            }
        }
        if ((lane & 15) == 0) {
            const int rbase = mt * 16 + (lane >> 4) * 8;
#pragma unroll
            for (int i = 0; i < 8; ++i) {
                atomicAdd(&red[rbase + i][0], py0[i]);
                atomicAdd(&red[rbase + i][1], py1[i]);
                atomicAdd(&red[rbase + i][2], pt0[i]);
                atomicAdd(&red[rbase + i][3], pt1[i]);
            }
        }
    }
    __syncthreads();

    // -------- final writeout ---------------------------------------------
    if (tid < MROWS) {
        const int r = r0 + tid;
        const float z0 = z[r * 2 + 0];
        const float z1 = z[r * 2 + 1];
        const float y0 = red[tid][0] + b3[0];
        const float y1 = red[tid][1] + b3[1];
        out[r * 2 + 0] = KFLOW * z0 + KFLOW * y0;
        out[r * 2 + 1] = KFLOW * z1 + KFLOW * y1;
        const float trace = KFLOW * 2.0f + KFLOW * (red[tid][2] + red[tid][3]);
        out[B_TOTAL * 2 + r] = -trace;
    }
}

// ---------------------------------------------------------------------------
extern "C" void kernel_launch(void* const* d_in, const int* in_sizes, int n_in,
                              void* d_out, int out_size, void* d_ws, size_t ws_size,
                              hipStream_t stream) {
    (void)in_sizes; (void)n_in; (void)out_size; (void)ws_size;
    const float* t  = (const float*)d_in[0];
    const float* z  = (const float*)d_in[1];
    const float* W1 = (const float*)d_in[2];
    const float* b1 = (const float*)d_in[3];
    const float* W2 = (const float*)d_in[4];
    const float* b2 = (const float*)d_in[5];
    const float* W3 = (const float*)d_in[6];
    const float* b3 = (const float*)d_in[7];
    unsigned short* W2bf = (unsigned short*)d_ws;   // 512 KB bf16 copy of W2

    cvt_w2_bf16<<<(INNER * INNER) / 256, 256, 0, stream>>>(W2, W2bf);
    pflow_kernel<<<B_TOTAL / MROWS, 256, 0, stream>>>(
        t, z, W1, b1, W2bf, b2, W3, b3, (float*)d_out);
}
